// KGRS_21457656611240
// MI455X (gfx1250) — matrix operand order repaired and checked by reading, not compiled, not run on hardware
//
#include <hip/hip_runtime.h>
#include <hip/hip_bf16.h>
#include <cstdint>

// Problem constants from the reference
#define BB   8192
#define NN   64
#define DD   64
#define HIDD 64
#define HH   2

typedef _Float16 v4h  __attribute__((ext_vector_type(4)));
typedef _Float16 v8h  __attribute__((ext_vector_type(8)));
typedef _Float16 v16h __attribute__((ext_vector_type(16)));
typedef float    v8f  __attribute__((ext_vector_type(8)));

// Pointer element type for the async builtin, matching the compiler's
// diagnostic: '__attribute__((__vector_size__(4 * sizeof(int)))) int *'
typedef int async_b128_t __attribute__((__vector_size__(16)));

// LDS strides (in elements). 88 halves = 176 B = 44 DWORDs:
//  - multiple of 16 B -> aligned ds_load_b128 fragments
//  - gcd(44,64)=4, cycle length 16 -> 16 lanes hit 16 distinct bank groups
#define XSTR 88

#if defined(__gfx1250__) && __has_builtin(__builtin_amdgcn_global_load_async_to_lds_b128)
#define USE_ASYNC 1
#else
#define USE_ASYNC 0
#endif

#if __has_builtin(__builtin_amdgcn_s_wait_asynccnt)
#define ASYNC_WAIT() __builtin_amdgcn_s_wait_asynccnt(0)
#else
#define ASYNC_WAIT() asm volatile("s_wait_asynccnt 0x0" ::: "memory")
#endif

__launch_bounds__(128)
__global__ void gat_wmma_kernel(const float* __restrict__ item,
                                const float* __restrict__ r,
                                const float* __restrict__ t,
                                const float* __restrict__ mask,
                                const float* __restrict__ W,
                                const float* __restrict__ bias,
                                const float* __restrict__ hvec,
                                float* __restrict__ out)
{
    __shared__ _Float16 x_lds[NN][XSTR];          // X = r*t, f16
    __shared__ _Float16 wt_lds[HH][HIDD][XSTR];   // W transposed: [h][hid][d], f16
    __shared__ float    t_lds[NN][DD];            // raw t tile, f32
    __shared__ float    score[HH][NN];
    __shared__ float    invtot[HH];
    __shared__ float    nb[HH][DD];

    const int bidx = blockIdx.x;
    const int tid  = threadIdx.x;
    const size_t rtBase = (size_t)bidx * NN * DD;

    // ---------------- Phase 1: stage t (async->LDS), X=r*t (f16), W^T (f16) --
#if USE_ASYNC
    // Kick off the DMA-style copy of the t tile into LDS: 128 threads x 8 x 16B.
    #pragma unroll
    for (int i = 0; i < 8; ++i) {
        int e   = (i * 128 + tid) * 4;
        int row = e >> 6;
        int col = e & 63;
        __builtin_amdgcn_global_load_async_to_lds_b128(
            (async_b128_t*)(t + rtBase + e),
            (async_b128_t*)(&t_lds[row][col]),
            0, 0);
    }
#endif

    // Overlap with async copy: r tile into registers (8 x float4 per thread)
    float4 rv[8];
#if USE_ASYNC
    #pragma unroll
    for (int i = 0; i < 8; ++i) {
        int e = (i * 128 + tid) * 4;
        rv[i] = *(const float4*)(r + rtBase + e);
    }
#endif

    // W: 2*64*64 = 8192 floats, 64 per thread, scattered transposed into LDS
    #pragma unroll
    for (int i = 0; i < 64; ++i) {
        int idx = i * 128 + tid;
        int h   = idx >> 12;
        int d   = (idx >> 6) & 63;
        int k   = idx & 63;
        wt_lds[h][k][d] = (_Float16)W[idx];
    }

#if USE_ASYNC
    ASYNC_WAIT();
    __syncthreads();
    // X = r * t, with t read back from LDS (single HBM read of t)
    #pragma unroll
    for (int i = 0; i < 8; ++i) {
        int e   = (i * 128 + tid) * 4;
        int row = e >> 6;
        int col = e & 63;
        float4 tv = *(const float4*)&t_lds[row][col];
        v4h xv;
        xv.x = (_Float16)(rv[i].x * tv.x);
        xv.y = (_Float16)(rv[i].y * tv.y);
        xv.z = (_Float16)(rv[i].z * tv.z);
        xv.w = (_Float16)(rv[i].w * tv.w);
        *(v4h*)&x_lds[row][col] = xv;
    }
#else
    (void)rv;
    #pragma unroll
    for (int i = 0; i < 8; ++i) {
        int e   = (i * 128 + tid) * 4;
        int row = e >> 6;
        int col = e & 63;
        float4 rv2 = *(const float4*)(r + rtBase + e);
        float4 tv  = *(const float4*)(t + rtBase + e);
        v4h xv;
        xv.x = (_Float16)(rv2.x * tv.x);
        xv.y = (_Float16)(rv2.y * tv.y);
        xv.z = (_Float16)(rv2.z * tv.z);
        xv.w = (_Float16)(rv2.w * tv.w);
        *(v4h*)&x_lds[row][col] = xv;
        *(float4*)&t_lds[row][col] = tv;
    }
#endif
    __syncthreads();

    // ---------------- Phase 2: WMMA GEMM + fused relu/hvec-dot/exp ----------
    const int lane    = tid & 31;
    const int wave    = tid >> 5;
    const int rowBase = wave << 4;                // 16-row strip of N per wave
    const int half16  = lane & 15;
    const int hiHalf  = (lane >= 16) ? 1 : 0;

    #pragma unroll
    for (int h = 0; h < HH; ++h) {
        v8f acc[4] = {};                          // 4 column tiles of 16x16 f32

        #pragma unroll
        for (int ks = 0; ks < 2; ++ks) {          // K = 64 in two 32-steps
            const int k0   = ks * 32;
            const int aRow = rowBase + half16;
            const int aK   = k0 + (hiHalf ? 8 : 0);
            v8h alo = *(const v8h*)&x_lds[aRow][aK];
            v8h ahi = *(const v8h*)&x_lds[aRow][aK + 16];
            v16h a  = __builtin_shufflevector(alo, ahi,
                        0,1,2,3,4,5,6,7,8,9,10,11,12,13,14,15);

            const int bK = k0 + (hiHalf ? 8 : 0);
            #pragma unroll
            for (int ct = 0; ct < 4; ++ct) {
                const int col = ct * 16 + half16;
                v8h blo = *(const v8h*)&wt_lds[h][col][bK];
                v8h bhi = *(const v8h*)&wt_lds[h][col][bK + 16];
                v16h bf = __builtin_shufflevector(blo, bhi,
                            0,1,2,3,4,5,6,7,8,9,10,11,12,13,14,15);
                acc[ct] = __builtin_amdgcn_wmma_f32_16x16x32_f16(
                            false, a, false, bf, (short)0, acc[ct],
                            false, false);
            }
        }

        // acc[ct][j] = S[rowBase + j (+8 if hiHalf)][ct*16 + half16]
        float rowpart[8] = {0.f,0.f,0.f,0.f,0.f,0.f,0.f,0.f};
        #pragma unroll
        for (int ct = 0; ct < 4; ++ct) {
            const int colIdx = ct * 16 + half16;
            const float bb = bias[h * HIDD + colIdx];
            const float hv = hvec[h * HIDD + colIdx];
            #pragma unroll
            for (int j = 0; j < 8; ++j) {
                float sv = acc[ct][j] + bb;
                sv = sv > 0.f ? sv : 0.f;          // relu
                rowpart[j] += sv * hv;             // dot with hvec (partial)
            }
        }
        // reduce over the 16 column lanes (stay within each 16-lane group)
        #pragma unroll
        for (int j = 0; j < 8; ++j) {
            #pragma unroll
            for (int m = 1; m <= 8; m <<= 1)
                rowpart[j] += __shfl_xor(rowpart[j], m, 32);
        }
        if (half16 == 0) {
            const int rOff = rowBase + (hiHalf ? 8 : 0);
            #pragma unroll
            for (int j = 0; j < 8; ++j) {
                const int n = rOff + j;
                score[h][n] = __expf(rowpart[j]) * mask[(size_t)bidx * NN + n];
            }
        }
    }
    __syncthreads();

    // ---------------- Phase 3: softmax denominators -------------------------
    if (tid < HH) {
        float s = 0.f;
        #pragma unroll 8
        for (int n = 0; n < NN; ++n) s += score[tid][n];
        invtot[tid] = 1.0f / s;
    }
    __syncthreads();

    // ---------------- Phase 4: nb[h] = (score/total)^T @ t ------------------
    {
        const int h = tid >> 6;                   // 0..1
        const int d = tid & 63;                   // 0..63
        float accd = 0.f;
        #pragma unroll 8
        for (int n = 0; n < NN; ++n)
            accd += score[h][n] * t_lds[n][d];
        nb[h][d] = accd * invtot[h];
    }
    __syncthreads();

    // ---------------- Phase 5: out = item + mean over heads -----------------
    if (tid < DD) {
        const size_t o = (size_t)bidx * DD + tid;
        out[o] = item[o] + 0.5f * (nb[0][tid] + nb[1][tid]);
    }
}

extern "C" void kernel_launch(void* const* d_in, const int* in_sizes, int n_in,
                              void* d_out, int out_size, void* d_ws, size_t ws_size,
                              hipStream_t stream) {
    const float* item = (const float*)d_in[0];   // (B, D)
    const float* r    = (const float*)d_in[1];   // (B, N, D)
    const float* t    = (const float*)d_in[2];   // (B, N, D)
    const float* mask = (const float*)d_in[3];   // (B, N)
    const float* W    = (const float*)d_in[4];   // (H, D, HID)
    const float* bias = (const float*)d_in[5];   // (H, HID)
    const float* hvec = (const float*)d_in[6];   // (H, HID)
    float* out        = (float*)d_out;           // (B, D)

    (void)in_sizes; (void)n_in; (void)out_size; (void)d_ws; (void)ws_size;

    gat_wmma_kernel<<<dim3(BB), dim3(128), 0, stream>>>(
        item, r, t, mask, W, bias, hvec, out);
}
